// Decoder_50508815401316
// MI455X (gfx1250) — matrix operand order repaired and checked by reading, not compiled, not run on hardware
//
#include <hip/hip_runtime.h>
#include <hip/hip_bf16.h>

// ---------------------------------------------------------------------------
// Decoder step: Bahdanau attention + GRU (h0==0 => recurrent GEMM vanishes)
// + vocab projection.  All large GEMMs run on v_wmma_f32_16x16x32_bf16.
// v2: pre-converted bf16 A operands (no per-iteration cvt VALU in GEMM loops),
//     dual N-tile per wave (2 independent WMMA chains, A-fragment reuse).
// ---------------------------------------------------------------------------

#define BATCH   64
#define SEQ     1024
#define UNITS   1024
#define EMB     256
#define VOCAB   32000
#define GRU_IN  (UNITS + EMB)     // 1280
#define GRU_OUT (3 * UNITS)       // 3072

typedef __attribute__((ext_vector_type(16))) __bf16 v16bf;
typedef __attribute__((ext_vector_type(8)))  float  v8f;

union ABf {
    v16bf          v;
    unsigned short h[16];
    unsigned int   u[8];
};

// fp32 -> bf16 bits, round-to-nearest-even (used only in one-time prep passes)
__device__ __forceinline__ unsigned short f2bf(float f) {
    unsigned u = __float_as_uint(f);
    unsigned r = u + 0x7FFFu + ((u >> 16) & 1u);
    return (unsigned short)(r >> 16);
}

__device__ __forceinline__ void fill8(ABf& d, uint4 a, uint4 b) {
    d.u[0] = a.x; d.u[1] = a.y; d.u[2] = a.z; d.u[3] = a.w;
    d.u[4] = b.x; d.u[5] = b.y; d.u[6] = b.z; d.u[7] = b.w;
}

// ---------------------------------------------------------------------------
// fp32 -> packed bf16 (row-major, same shape).  count2 = #elements / 2.
// ---------------------------------------------------------------------------
__global__ void to_bf16(const float* __restrict__ src,
                        unsigned int* __restrict__ dst, int count2) {
    int i = blockIdx.x * blockDim.x + threadIdx.x;
    if (i < count2) {
        float lo = src[2 * i], hi = src[2 * i + 1];
        dst[i] = (unsigned)f2bf(lo) | ((unsigned)f2bf(hi) << 16);
    }
}

// ---------------------------------------------------------------------------
// W1 (UNITS x UNITS fp32, row-major) -> W1t bf16, [n][k] layout
// (k contiguous per output column, i.e. WMMA-B friendly).
// ---------------------------------------------------------------------------
__global__ void w1_transpose_bf16(const float* __restrict__ W1,
                                  unsigned short* __restrict__ W1t) {
    __shared__ float tile[32][33];
    int tx = threadIdx.x, ty = threadIdx.y;   // 32 x 8
    int bx = blockIdx.x,  by = blockIdx.y;
#pragma unroll
    for (int i = 0; i < 4; ++i) {
        int k = by * 32 + ty + i * 8;
        int n = bx * 32 + tx;
        tile[ty + i * 8][tx] = W1[(size_t)k * UNITS + n];
    }
    __syncthreads();
#pragma unroll
    for (int i = 0; i < 4; ++i) {
        int n = bx * 32 + ty + i * 8;
        int k = by * 32 + tx;
        W1t[(size_t)n * UNITS + k] = f2bf(tile[tx][ty + i * 8]);
    }
}

// ---------------------------------------------------------------------------
// D[64][N] = A[64][K] @ B[K][N] + bias[N].
// A: pre-converted bf16 row-major.  B: fp32, transposed+converted through LDS.
// Block: 128 threads = 4 waves; wave w -> rows 16w..16w+15; block covers a
// 32-wide N tile (two 16-tiles per wave sharing one A fragment).
// ---------------------------------------------------------------------------
__global__ void gemm64_bf16(const unsigned short* __restrict__ Abf,
                            const float* __restrict__ Bm,
                            const float* __restrict__ bias,
                            float* __restrict__ D,
                            int K, int N, int KPAD) {
    extern __shared__ unsigned short Bs[];          // [32][KPAD] bf16
    const int tid = threadIdx.x;
    const int n0  = blockIdx.x * 32;

    // stage B tile (K x 32 fp32) into LDS as bf16, k-contiguous per column
    for (int idx = tid; idx < (K << 5); idx += 128) {
        int k = idx >> 5, n = idx & 31;
        Bs[n * KPAD + k] = f2bf(Bm[(size_t)k * N + n0 + n]);
    }
    __syncthreads();

    const int wave = tid >> 5, lane = tid & 31, hi = lane >> 4;
    const int mrow = wave * 16 + (lane & 15);
    const unsigned short* ap  = Abf + (size_t)mrow * K + (hi << 4);
    const unsigned short* bp0 = Bs + (lane & 15) * KPAD + (hi << 4);
    const unsigned short* bp1 = bp0 + 16 * KPAD;

    v8f c0 = {}, c1 = {};
    const int nsteps = K >> 5;
    for (int kk = 0; kk < nsteps; ++kk) {
        ABf a, fb0, fb1;
        fill8(a, *(const uint4*)(ap + kk * 32), *(const uint4*)(ap + kk * 32 + 8));
        const uint4* q0 = (const uint4*)(bp0 + kk * 32);
        fill8(fb0, q0[0], q0[1]);
        const uint4* q1 = (const uint4*)(bp1 + kk * 32);
        fill8(fb1, q1[0], q1[1]);
        c0 = __builtin_amdgcn_wmma_f32_16x16x32_bf16(false, a.v, false, fb0.v,
                                                     (short)0, c0, false, false);
        c1 = __builtin_amdgcn_wmma_f32_16x16x32_bf16(false, a.v, false, fb1.v,
                                                     (short)0, c1, false, false);
    }

    const int   n     = n0 + (lane & 15);
    const float bb0   = bias[n];
    const float bb1   = bias[n + 16];
    const int   rbase = wave * 16 + 8 * hi;
#pragma unroll
    for (int v = 0; v < 8; ++v) {
        D[(size_t)(rbase + v) * N + n]      = c0[v] + bb0;
        D[(size_t)(rbase + v) * N + n + 16] = c1[v] + bb1;
    }
}

// ---------------------------------------------------------------------------
// Attention scores: score[b,s] = tanh(enc[b,s,:]@W1 + b1 + q[b,:]) . V + bV
// Grid (S/16, B); block 256 = 8 waves.  enc 16-row tile staged to LDS (bf16,
// one-time cvt); each wave sweeps 4 pairs of N-tiles (t, t+8), fusing
// tanh + dot-with-V into the WMMA epilogue; partials reduced via ds_add_f32.
// ---------------------------------------------------------------------------
#define ASTRIDE 1032   // 1024 + 8 pad (bank-conflict-free row stride)

__global__ void attn_score(const float* __restrict__ enc,
                           const unsigned short* __restrict__ W1t,
                           const float* __restrict__ b1,
                           const float* __restrict__ q,
                           const float* __restrict__ Vv,
                           const float* __restrict__ bV,
                           float* __restrict__ scores) {
    __shared__ unsigned short As[16 * ASTRIDE];
    __shared__ float sc[16];
    const int tid = threadIdx.x;
    const int s0  = blockIdx.x * 16;
    const int b   = blockIdx.y;

    const size_t encBase = ((size_t)b * SEQ + s0) * UNITS;
    for (int idx = tid; idx < 16 * UNITS; idx += 256) {
        int r = idx >> 10, k = idx & 1023;
        As[r * ASTRIDE + k] = f2bf(enc[encBase + (size_t)r * UNITS + k]);
    }
    if (tid < 16) sc[tid] = 0.f;
    __syncthreads();

    const int wave = tid >> 5, lane = tid & 31, hi = lane >> 4;
    const unsigned short* arow = As + (lane & 15) * ASTRIDE + (hi << 4);

    float rowsum[8];
#pragma unroll
    for (int v = 0; v < 8; ++v) rowsum[v] = 0.f;

#pragma unroll 1
    for (int j = 0; j < 4; ++j) {
        const int n0 = (wave + 16 * j) * 16 + (lane & 15);   // tile t
        const int n1 = n0 + 128;                             // tile t+8
        const unsigned short* bcol0 = W1t + (size_t)n0 * UNITS + (hi << 4);
        const unsigned short* bcol1 = W1t + (size_t)n1 * UNITS + (hi << 4);
        v8f c0 = {}, c1 = {};
        for (int kk = 0; kk < UNITS / 32; ++kk) {
            ABf a, fb0, fb1;
            fill8(a, *(const uint4*)(arow + kk * 32),
                     *(const uint4*)(arow + kk * 32 + 8));
            const uint4* q0 = (const uint4*)(bcol0 + kk * 32);
            fill8(fb0, q0[0], q0[1]);
            const uint4* q1 = (const uint4*)(bcol1 + kk * 32);
            fill8(fb1, q1[0], q1[1]);
            c0 = __builtin_amdgcn_wmma_f32_16x16x32_bf16(false, a.v, false, fb0.v,
                                                         (short)0, c0, false, false);
            c1 = __builtin_amdgcn_wmma_f32_16x16x32_bf16(false, a.v, false, fb1.v,
                                                         (short)0, c1, false, false);
        }
        const float add0 = b1[n0] + q[b * UNITS + n0];
        const float add1 = b1[n1] + q[b * UNITS + n1];
        const float vn0  = Vv[n0];
        const float vn1  = Vv[n1];
#pragma unroll
        for (int v = 0; v < 8; ++v) {
            rowsum[v] += tanhf(c0[v] + add0) * vn0;
            rowsum[v] += tanhf(c1[v] + add1) * vn1;
        }
    }

    const int rbase = hi * 8;
#pragma unroll
    for (int v = 0; v < 8; ++v)
        atomicAdd(&sc[rbase + v], rowsum[v]);   // ds_add_f32
    __syncthreads();
    if (tid < 16)
        scores[b * SEQ + s0 + tid] = sc[tid] + bV[0];
}

// ---------------------------------------------------------------------------
// Softmax over S per batch row, context = attn^T @ enc, assemble
// xt = [context | emb[x[b]]].  One block per b, 256 threads.
// ---------------------------------------------------------------------------
__global__ void ctx_kernel(const float* __restrict__ scores,
                           const float* __restrict__ enc,
                           const float* __restrict__ emb,
                           const int* __restrict__ x,
                           float* __restrict__ xt,
                           float* __restrict__ attn_out) {
    __shared__ float sa[SEQ];
    __shared__ float red[256];
    const int b = blockIdx.x, tid = threadIdx.x;

    float lm = -1e30f;
#pragma unroll
    for (int i = 0; i < 4; ++i) {
        float v = scores[b * SEQ + tid + 256 * i];
        sa[tid + 256 * i] = v;
        lm = fmaxf(lm, v);
    }
    red[tid] = lm;
    __syncthreads();
    for (int s = 128; s > 0; s >>= 1) {
        if (tid < s) red[tid] = fmaxf(red[tid], red[tid + s]);
        __syncthreads();
    }
    const float m = red[0];
    __syncthreads();

    float ls = 0.f;
#pragma unroll
    for (int i = 0; i < 4; ++i) ls += expf(sa[tid + 256 * i] - m);
    red[tid] = ls;
    __syncthreads();
    for (int s = 128; s > 0; s >>= 1) {
        if (tid < s) red[tid] += red[tid + s];
        __syncthreads();
    }
    const float inv = 1.0f / red[0];
    __syncthreads();

#pragma unroll
    for (int i = 0; i < 4; ++i) {
        int idx = tid + 256 * i;
        float a = expf(sa[idx] - m) * inv;
        sa[idx] = a;
        attn_out[b * SEQ + idx] = a;
    }
    __syncthreads();

    float a0 = 0.f, a1 = 0.f, a2 = 0.f, a3 = 0.f;
    for (int s = 0; s < SEQ; ++s) {
        float a = sa[s];
        const float* ep = enc + ((size_t)(b * SEQ + s)) * UNITS + tid;
        a0 += a * ep[0];
        a1 += a * ep[256];
        a2 += a * ep[512];
        a3 += a * ep[768];
    }
    float* xrow = xt + (size_t)b * GRU_IN;
    xrow[tid]       = a0;
    xrow[tid + 256] = a1;
    xrow[tid + 512] = a2;
    xrow[tid + 768] = a3;

    const int tok = x[b];
    xrow[UNITS + tid] = emb[(size_t)tok * EMB + tid];   // EMB == blockDim
}

// ---------------------------------------------------------------------------
// GRU gates.  h0 == 0 => hr = gru_bias[1];  h = (1-z)*hh.
// xi already includes gru_bias[0] from the GEMM epilogue.
// ---------------------------------------------------------------------------
__global__ void gru_gates(const float* __restrict__ xi,
                          const float* __restrict__ gbias,
                          float* __restrict__ hout) {
    const int idx = blockIdx.x * blockDim.x + threadIdx.x;   // 65536
    const int b = idx >> 10, j = idx & 1023;
    const float* xr = xi + (size_t)b * GRU_OUT;
    const float  xz = xr[j], xrr = xr[UNITS + j], xh = xr[2 * UNITS + j];
    const float* hb = gbias + GRU_OUT;                       // bias row 1
    const float z  = 1.f / (1.f + expf(-(xz  + hb[j])));
    const float r  = 1.f / (1.f + expf(-(xrr + hb[UNITS + j])));
    const float hh = tanhf(xh + r * hb[2 * UNITS + j]);
    hout[idx] = (1.f - z) * hh;
}

// ---------------------------------------------------------------------------
extern "C" void kernel_launch(void* const* d_in, const int* in_sizes, int n_in,
                              void* d_out, int out_size, void* d_ws, size_t ws_size,
                              hipStream_t stream) {
    (void)in_sizes; (void)n_in; (void)out_size; (void)ws_size;

    const int*   x      = (const int*)  d_in[0];
    const float* hidden = (const float*)d_in[1];
    const float* enc    = (const float*)d_in[2];
    const float* W1     = (const float*)d_in[3];
    const float* b1     = (const float*)d_in[4];
    const float* W2     = (const float*)d_in[5];
    const float* b2     = (const float*)d_in[6];
    const float* Vv     = (const float*)d_in[7];
    const float* bV     = (const float*)d_in[8];
    const float* emb    = (const float*)d_in[9];
    const float* gk     = (const float*)d_in[10];
    /* d_in[11] = gru_rkernel: unused, h0 == 0 in the reference */
    const float* gbias  = (const float*)d_in[12];
    const float* fcW    = (const float*)d_in[13];
    const float* fcb    = (const float*)d_in[14];

    char* ws = (char*)d_ws;
    size_t off = 0;
    unsigned short* W1t   = (unsigned short*)(ws + off); off += (size_t)UNITS * UNITS * 2;  // 2 MB
    float* q              = (float*)(ws + off);          off += (size_t)BATCH * UNITS * 4;
    float* scores         = (float*)(ws + off);          off += (size_t)BATCH * SEQ * 4;
    float* xt             = (float*)(ws + off);          off += (size_t)BATCH * GRU_IN * 4;
    float* xi             = (float*)(ws + off);          off += (size_t)BATCH * GRU_OUT * 4;
    unsigned short* hidb  = (unsigned short*)(ws + off); off += (size_t)BATCH * UNITS * 2;
    unsigned short* xtb   = (unsigned short*)(ws + off); off += (size_t)BATCH * GRU_IN * 2;
    unsigned short* hb    = (unsigned short*)(ws + off); off += (size_t)BATCH * UNITS * 2;

    float* logits   = (float*)d_out;                                 // 64x32000
    float* hout     = logits + (size_t)BATCH * VOCAB;                // 64x1024
    float* attn_out = hout + (size_t)BATCH * UNITS;                  // 64x1024

    // 0) W1 -> bf16 column-major (WMMA-B layout), reused by 4096 tiles via L2
    w1_transpose_bf16<<<dim3(UNITS / 32, UNITS / 32), dim3(32, 8), 0, stream>>>(W1, W1t);

    // 1) q = hidden @ W2 + b2
    to_bf16<<<(BATCH * UNITS / 2 + 255) / 256, 256, 0, stream>>>(
        hidden, (unsigned int*)hidb, BATCH * UNITS / 2);
    gemm64_bf16<<<UNITS / 32, 128, 32 * (UNITS + 8) * 2, stream>>>(
        hidb, W2, b2, q, UNITS, UNITS, UNITS + 8);

    // 2) scores = tanh(enc@W1 + b1 + q) . V + bV     (dominant: ~137 GF, WMMA)
    attn_score<<<dim3(SEQ / 16, BATCH), 256, 0, stream>>>(
        enc, W1t, b1, q, Vv, bV, scores);

    // 3) softmax + context + xt = [context | emb[x]]
    ctx_kernel<<<BATCH, 256, 0, stream>>>(scores, enc, emb, x, xt, attn_out);

    // 4) xi = xt @ gru_kernel + gru_bias[0]
    to_bf16<<<(BATCH * GRU_IN / 2 + 255) / 256, 256, 0, stream>>>(
        xt, (unsigned int*)xtb, BATCH * GRU_IN / 2);
    gemm64_bf16<<<GRU_OUT / 32, 128, 32 * (GRU_IN + 8) * 2, stream>>>(
        xtb, gk, gbias, xi, GRU_IN, GRU_OUT, GRU_IN + 8);

    // 5) gates -> h
    gru_gates<<<(BATCH * UNITS) / 256, 256, 0, stream>>>(xi, gbias, hout);

    // 6) logits = h @ fc_W + fc_b
    to_bf16<<<(BATCH * UNITS / 2 + 255) / 256, 256, 0, stream>>>(
        hout, (unsigned int*)hb, BATCH * UNITS / 2);
    gemm64_bf16<<<VOCAB / 32, 128, 32 * (UNITS + 8) * 2, stream>>>(
        hb, fcW, fcb, logits, UNITS, VOCAB, UNITS + 8);
}
